// PNAPCSAFT2_9577777070401
// MI455X (gfx1250) — matrix operand order repaired and checked by reading, not compiled, not run on hardware
//
#include <hip/hip_runtime.h>
#include <math.h>

// ---------------------------------------------------------------------------
// PNA GNN forward for MI455X (gfx1250, wave32, WMMA bf16).
// All H=64 GEMMs run on v_wmma_f32_16x16x32_bf16 with fp32 accumulation.
// Edge-encoder GEMM and post/lin GEMMs are algebraically composed away.
// Segment aggregation via L2 atomics (add + inline-asm min_num/max_num f32).
// Edge-kernel weights staged to LDS via async global->LDS copy (ASYNCcnt).
// ---------------------------------------------------------------------------

#define NN 50000
#define NE 800000
#define NG 1024

typedef __attribute__((ext_vector_type(16))) __bf16 bf16x16;
typedef __attribute__((ext_vector_type(8)))  float  f32x8;
typedef int gvec4 __attribute__((__vector_size__(16)));   // int4 for async-LDS builtin

union Frag { uint4 u[2]; bf16x16 v; };

#if defined(__has_builtin)
#if __has_builtin(__builtin_amdgcn_global_load_async_to_lds_b128) && \
    __has_builtin(__builtin_amdgcn_s_wait_asynccnt)
#define USE_ASYNC_LDS 1
#endif
#endif

__device__ __forceinline__ unsigned short f2bf(float f) {
  unsigned u = __float_as_uint(f);
  unsigned r = (u >> 16) & 1u;
  return (unsigned short)((u + 0x7fffu + r) >> 16);
}
__device__ __forceinline__ float bf2f(unsigned short h) {
  return __uint_as_float(((unsigned)h) << 16);
}

__device__ __forceinline__ f32x8 wmma_bf(bf16x16 a, bf16x16 b, f32x8 c) {
  // D = A(16x32 bf16) x B(32x16 bf16) + C(16x16 f32)
  return __builtin_amdgcn_wmma_f32_16x16x32_bf16(false, a, false, b, (short)0, c,
                                                 false, false);
}

// CDNA5 float min/max atomics (ISA 15.18.3 opcodes 81/82), no return needed.
__device__ __forceinline__ void atomic_fmin(float* p, float v) {
  asm volatile("global_atomic_min_num_f32 %0, %1, off" :: "v"(p), "v"(v) : "memory");
}
__device__ __forceinline__ void atomic_fmax(float* p, float v) {
  asm volatile("global_atomic_max_num_f32 %0, %1, off" :: "v"(p), "v"(v) : "memory");
}

// B fragment from global: packed layout [tile][lane(32)][16 halves]
__device__ __forceinline__ bf16x16 loadB(const unsigned short* __restrict__ base,
                                         int tile, int lane) {
  Frag b;
  const uint4* p = (const uint4*)(base + ((size_t)tile * 32 + lane) * 16);
  b.u[0] = p[0]; b.u[1] = p[1];
  return b.v;
}
// A fragment: row-major bf16 row; lane<16 holds K {off..off+7, off+16..off+23}
__device__ __forceinline__ void loadA(Frag& a, const unsigned short* row, int off) {
  a.u[0] = *(const uint4*)(row + off);
  a.u[1] = *(const uint4*)(row + off + 16);
}

// --------------------------- encoders --------------------------------------

__global__ void k_node_enc(const float* __restrict__ x, const float* __restrict__ w,
                           const float* __restrict__ b, unsigned short* __restrict__ hbf) {
  __shared__ float sw[32 * 64];
  __shared__ float sb[64];
  for (int i = threadIdx.x; i < 32 * 64; i += blockDim.x) sw[i] = w[i];
  for (int i = threadIdx.x; i < 64; i += blockDim.x) sb[i] = b[i];
  __syncthreads();
  int n = blockIdx.x * blockDim.x + threadIdx.x;
  if (n >= NN) return;
  float xi[32];
  #pragma unroll
  for (int k = 0; k < 32; ++k) xi[k] = x[(size_t)n * 32 + k];
  for (int j = 0; j < 64; ++j) {
    float a = sb[j];
    #pragma unroll
    for (int k = 0; k < 32; ++k) a = fmaf(xi[k], sw[k * 64 + j], a);
    a = a > 0.f ? a : 0.01f * a;                 // leaky_relu
    hbf[(size_t)n * 64 + j] = f2bf(a);
  }
}

__global__ void k_edge_enc(const float* __restrict__ ea, const float* __restrict__ w,
                           const float* __restrict__ b, unsigned short* __restrict__ ebf) {
  __shared__ float sw[11 * 64];
  __shared__ float sb[64];
  for (int i = threadIdx.x; i < 11 * 64; i += blockDim.x) sw[i] = w[i];
  for (int i = threadIdx.x; i < 64; i += blockDim.x) sb[i] = b[i];
  __syncthreads();
  int e = blockIdx.x * blockDim.x + threadIdx.x;
  if (e >= NE) return;
  float xi[11];
  #pragma unroll
  for (int k = 0; k < 11; ++k) xi[k] = ea[(size_t)e * 11 + k];
  for (int j = 0; j < 64; ++j) {
    float a = sb[j];
    #pragma unroll
    for (int k = 0; k < 11; ++k) a = fmaf(xi[k], sw[k * 64 + j], a);
    a = a > 0.f ? a : 0.01f * a;
    ebf[(size_t)e * 64 + j] = f2bf(a);
  }
}

// --------------------------- degree / amp / att ----------------------------

__global__ void k_deg(const int* __restrict__ dstI, float* __restrict__ deg) {
  int e = blockIdx.x * blockDim.x + threadIdx.x;
  if (e < NE) atomicAdd(deg + dstI[e], 1.0f);
}

__global__ void k_ampatt(const float* __restrict__ deg, float* __restrict__ amp,
                         float* __restrict__ att, float avgLog) {
  int n = blockIdx.x * blockDim.x + threadIdx.x;
  if (n >= NN) return;
  float d = deg[n];
  float dc = d < 1.f ? 1.f : d;
  float l = logf(dc + 1.f);
  amp[n] = l / avgLog;
  att[n] = avgLog / l;
}

__global__ void k_fill(float* __restrict__ p, float v, int n) {
  int t = blockIdx.x * blockDim.x + threadIdx.x;
  if (t < n) p[t] = v;
}

// --------------------------- weight composition ----------------------------

// We = enc_w @ pre_w[128:192], bias' = pre_b + enc_b @ pre_w[128:192]
__global__ void k_compose_pre(const float* __restrict__ encw, const float* __restrict__ encb,
                              const float* __restrict__ prew, const float* __restrict__ preb,
                              float* __restrict__ We, float* __restrict__ biasOut) {
  int t = blockIdx.x * blockDim.x + threadIdx.x;
  if (t < 64 * 64) {
    int k = t >> 6, n = t & 63;
    float a = 0.f;
    for (int j = 0; j < 64; ++j) a = fmaf(encw[k * 64 + j], prew[(128 + j) * 64 + n], a);
    We[t] = a;
  }
  if (t < 64) {
    float a = preb[t];
    for (int j = 0; j < 64; ++j) a = fmaf(encb[j], prew[(128 + j) * 64 + t], a);
    biasOut[t] = a;
  }
}

// P = post_w @ lin_w (832x64), b2 = post_b @ lin_w + lin_b
__global__ void k_compose_post(const float* __restrict__ postw, const float* __restrict__ postb,
                               const float* __restrict__ linw, const float* __restrict__ linb,
                               float* __restrict__ P, float* __restrict__ b2) {
  int t = blockIdx.x * blockDim.x + threadIdx.x;
  if (t < 832 * 64) {
    int k = t >> 6, n = t & 63;
    float a = 0.f;
    for (int j = 0; j < 64; ++j) a = fmaf(postw[k * 64 + j], linw[j * 64 + n], a);
    P[t] = a;
  }
  if (t < 64) {
    float a = linb[t];
    for (int j = 0; j < 64; ++j) a = fmaf(postb[j], linw[j * 64 + t], a);
    b2[t] = a;
  }
}

// Pack f32 [K][64] weights into WMMA B layout: [nt][kt][lane][16] bf16.
// lane<16: column N=nt*16+lane, K = kt*32 + {0..7, 16..23}; lane>=16: K += 8.
__global__ void k_pack_b(const float* __restrict__ src, unsigned short* __restrict__ dst, int KT) {
  int t = blockIdx.x * blockDim.x + threadIdx.x;
  if (t >= 4 * KT * 32) return;
  int lane = t & 31;
  int kt = (t >> 5) % KT;
  int nt = (t >> 5) / KT;
  int n = nt * 16 + (lane & 15);
  int kb = kt * 32 + ((lane >> 4) << 3);
  unsigned short* o = dst + ((size_t)(nt * KT + kt) * 32 + lane) * 16;
  #pragma unroll
  for (int j = 0; j < 16; ++j) o[j] = f2bf(src[(size_t)(kb + j + ((j >> 3) << 3)) * 64 + n]);
}

// --------------------------- edge message + scatter ------------------------
// Per wave: 16 edges. m = h[dst]@W1 + h[src]@W2 + e@We + b'   (all via WMMA)
// then atomic scatter of sum / sum^2 / min / max keyed by dst node.
// Packed weights (24 KB) staged to LDS once per workgroup.
// Loop order: (s,kt) outer / nt inner -> one live A fragment + 4 independent
// C accumulators (no spills, better XDL pipelining than a C-chained order).
__global__ void k_msg(const unsigned short* __restrict__ hbf,
                      const unsigned short* __restrict__ ebf,
                      const int* __restrict__ srcI, const int* __restrict__ dstI,
                      const unsigned short* __restrict__ packW,  // [3 src][nt4][kt2][32][16]
                      const float* __restrict__ bias,
                      float* __restrict__ sum, float* __restrict__ ssq,
                      float* __restrict__ smn, float* __restrict__ smx) {
  __shared__ unsigned short sW[3 * 4096];                 // 24 KB of B fragments
  {
    uint4* sw4 = (uint4*)sW;
    const uint4* gw4 = (const uint4*)packW;
#ifdef USE_ASYNC_LDS
    for (int i = threadIdx.x; i < 1536; i += blockDim.x) {
      __builtin_amdgcn_global_load_async_to_lds_b128(
          (__attribute__((address_space(1))) gvec4*)(gw4 + i),
          (__attribute__((address_space(3))) gvec4*)(sw4 + i), 0, 0);
    }
    __builtin_amdgcn_s_wait_asynccnt(0);
#else
    for (int i = threadIdx.x; i < 1536; i += blockDim.x) sw4[i] = gw4[i];
#endif
  }
  __syncthreads();

  int wave = (blockIdx.x * blockDim.x + threadIdx.x) >> 5;
  int lane = threadIdx.x & 31;
  int e0 = wave << 4;
  if (e0 >= NE) return;
  int myRow = lane & 15;
  int dn = dstI[e0 + myRow];
  int sn = srcI[e0 + myRow];
  __builtin_prefetch(dstI + e0 + 2048, 0, 0);    // global_prefetch_b8 ahead
  int khalf = (lane >> 4) << 3;                  // 0 or 8

  const unsigned short* rowp[3];
  rowp[0] = hbf + (size_t)dn * 64;               // h[dst]
  rowp[1] = hbf + (size_t)sn * 64;               // h[src]
  rowp[2] = ebf + (size_t)(e0 + myRow) * 64;     // e

  f32x8 c[4];
  #pragma unroll
  for (int nt = 0; nt < 4; ++nt) c[nt] = (f32x8){0.f, 0.f, 0.f, 0.f, 0.f, 0.f, 0.f, 0.f};

  #pragma unroll
  for (int s = 0; s < 3; ++s)
    #pragma unroll
    for (int kt = 0; kt < 2; ++kt) {
      Frag a;
      loadA(a, rowp[s], kt * 32 + khalf);
      #pragma unroll
      for (int nt = 0; nt < 4; ++nt) {
        // B fragment from LDS (ds_load_b128)
        Frag b;
        const uint4* p = (const uint4*)(sW + ((size_t)(s * 8 + nt * 2 + kt) * 32 + lane) * 16);
        b.u[0] = p[0]; b.u[1] = p[1];
        c[nt] = wmma_bf(a.v, b.v, c[nt]);
      }
    }

  int col0 = lane & 15;
  int rbase = (lane >> 4) << 3;
  #pragma unroll
  for (int nt = 0; nt < 4; ++nt) {
    int f = nt * 16 + col0;
    float bv = bias[f];
    #pragma unroll
    for (int r = 0; r < 8; ++r) {
      int rr = rbase + r;                        // edge row in tile (C layout)
      int node = __shfl(dn, rr, 32);
      float v = c[nt][r] + bv;
      size_t off = (size_t)node * 64 + f;
      atomicAdd(sum + off, v);
      atomicAdd(ssq + off, v * v);
      atomic_fmin(smn + off, v);
      atomic_fmax(smx + off, v);
    }
  }
}

// --------------------------- finalize aggregates ---------------------------

__global__ void k_agg_fin(const float* __restrict__ sum, const float* __restrict__ ssq,
                          const float* __restrict__ smn, const float* __restrict__ smx,
                          const float* __restrict__ deg, unsigned short* __restrict__ aggbf) {
  int t = blockIdx.x * blockDim.x + threadIdx.x;
  if (t >= NN * 64) return;
  int n = t >> 6, f = t & 63;
  float d = deg[n];
  float dc = d < 1.f ? 1.f : d;
  float mean = sum[t] / dc;
  float msq  = ssq[t] / dc;
  float var = msq - mean * mean;
  if (var < 0.f) var = 0.f;
  float sd = sqrtf(var + 1e-5f);
  bool has = d > 0.f;
  float mn = has ? smn[t] : 0.f;
  float mx = has ? smx[t] : 0.f;
  unsigned short* o = aggbf + (size_t)n * 256;
  o[f]        = f2bf(mean);
  o[64 + f]   = f2bf(mn);
  o[128 + f]  = f2bf(mx);
  o[192 + f]  = f2bf(sd);
}

// --------------------------- node update GEMM ------------------------------
// out = h@Ph + agg@P1 + amp*(agg@P2) + att*(agg@P3) + b2 ; BN stats via atomics
// Single live C accumulator at a time (amp/att paths folded into scalars).
__global__ void k_node_gemm(const unsigned short* __restrict__ hbf,
                            const unsigned short* __restrict__ aggbf,
                            const float* __restrict__ amp, const float* __restrict__ att,
                            const unsigned short* __restrict__ packPh,   // [nt4][kt2][32][16]
                            const unsigned short* __restrict__ packP1,   // [nt4][kt8][32][16]
                            const unsigned short* __restrict__ packP2,
                            const unsigned short* __restrict__ packP3,
                            const float* __restrict__ b2, float* __restrict__ outf,
                            float* __restrict__ bnsum, float* __restrict__ bnsq) {
  int wave = (blockIdx.x * blockDim.x + threadIdx.x) >> 5;
  int lane = threadIdx.x & 31;
  int n0 = wave << 4;
  if (n0 >= NN) return;
  int myRow = lane & 15;
  int khalf = (lane >> 4) << 3;

  Frag ag[8];
  const unsigned short* pa = aggbf + (size_t)(n0 + myRow) * 256;
  #pragma unroll
  for (int kt = 0; kt < 8; ++kt) loadA(ag[kt], pa, kt * 32 + khalf);
  const unsigned short* ph = hbf + (size_t)(n0 + myRow) * 64;

  float myAmp = amp[n0 + myRow];
  float myAtt = att[n0 + myRow];
  int col0 = lane & 15;
  int rbase = (lane >> 4) << 3;
  float ampR[8], attR[8];
  #pragma unroll
  for (int r = 0; r < 8; ++r) {
    ampR[r] = __shfl(myAmp, rbase + r, 32);
    attR[r] = __shfl(myAtt, rbase + r, 32);
  }

  for (int nt = 0; nt < 4; ++nt) {
    float acc[8];
    // amp * (agg @ P2)
    f32x8 c = {0.f, 0.f, 0.f, 0.f, 0.f, 0.f, 0.f, 0.f};
    #pragma unroll
    for (int kt = 0; kt < 8; ++kt) c = wmma_bf(ag[kt].v, loadB(packP2, nt * 8 + kt, lane), c);
    #pragma unroll
    for (int r = 0; r < 8; ++r) acc[r] = ampR[r] * c[r];
    // att * (agg @ P3), reusing the same accumulator registers
    c = (f32x8){0.f, 0.f, 0.f, 0.f, 0.f, 0.f, 0.f, 0.f};
    #pragma unroll
    for (int kt = 0; kt < 8; ++kt) c = wmma_bf(ag[kt].v, loadB(packP3, nt * 8 + kt, lane), c);
    #pragma unroll
    for (int r = 0; r < 8; ++r) acc[r] += attR[r] * c[r];
    // h @ Ph + agg @ P1  (h fragments reloaded per nt: L0-resident, saves VGPRs)
    c = (f32x8){0.f, 0.f, 0.f, 0.f, 0.f, 0.f, 0.f, 0.f};
    #pragma unroll
    for (int kt = 0; kt < 2; ++kt) {
      Frag ah;
      loadA(ah, ph, kt * 32 + khalf);
      c = wmma_bf(ah.v, loadB(packPh, nt * 2 + kt, lane), c);
    }
    #pragma unroll
    for (int kt = 0; kt < 8; ++kt) c = wmma_bf(ag[kt].v, loadB(packP1, nt * 8 + kt, lane), c);

    int f = nt * 16 + col0;
    float bb = b2[f];
    float ls = 0.f, lq = 0.f;
    #pragma unroll
    for (int r = 0; r < 8; ++r) {
      float v = acc[r] + c[r] + bb;
      outf[(size_t)(n0 + rbase + r) * 64 + f] = v;
      ls += v;
      lq += v * v;
    }
    atomicAdd(bnsum + f, ls);
    atomicAdd(bnsq + f, lq);
  }
}

__global__ void k_bn_apply(const float* __restrict__ outf, const float* __restrict__ bnsum,
                           const float* __restrict__ bnsq, const float* __restrict__ g,
                           const float* __restrict__ b, unsigned short* __restrict__ hbf) {
  int t = blockIdx.x * blockDim.x + threadIdx.x;
  if (t >= NN * 64) return;
  int f = t & 63;
  const float inv = 1.0f / (float)NN;
  float mu = bnsum[f] * inv;
  float var = bnsq[f] * inv - mu * mu;
  if (var < 0.f) var = 0.f;
  float v = g[f] * (outf[t] - mu) * rsqrtf(var + 1e-5f) + b[f];
  v = v > 0.f ? v : 0.f;
  hbf[t] = f2bf(v);
}

// --------------------------- pool + head -----------------------------------

__global__ void k_pool(const unsigned short* __restrict__ hbf, const int* __restrict__ batch,
                       float* __restrict__ pool) {
  int t = blockIdx.x * blockDim.x + threadIdx.x;
  if (t >= NN * 64) return;
  int n = t >> 6;
  atomicAdd(pool + (size_t)batch[n] * 64 + (t & 63), bf2f(hbf[t]));
}

__global__ void k_head_gemm(const float* __restrict__ in, const float* __restrict__ w,
                            const float* __restrict__ bias, float* __restrict__ out,
                            int Kin, int Kout, float* __restrict__ stat) {
  int t = blockIdx.x * blockDim.x + threadIdx.x;
  if (t >= NG * Kout) return;
  int row = t / Kout, j = t % Kout;
  const float* r = in + (size_t)row * Kin;
  float a = bias[j];
  for (int k = 0; k < Kin; ++k) a = fmaf(r[k], w[k * Kout + j], a);
  out[(size_t)row * Kout + j] = a;
  if (stat) { atomicAdd(stat + j, a); atomicAdd(stat + Kout + j, a * a); }
}

__global__ void k_head_bn(float* __restrict__ x, const float* __restrict__ g,
                          const float* __restrict__ b, int Kout,
                          const float* __restrict__ stat) {
  int t = blockIdx.x * blockDim.x + threadIdx.x;
  if (t >= NG * Kout) return;
  int j = t % Kout;
  const float inv = 1.0f / (float)NG;
  float mu = stat[j] * inv;
  float var = stat[Kout + j] * inv - mu * mu;
  if (var < 0.f) var = 0.f;
  float v = g[j] * (x[t] - mu) * rsqrtf(var + 1e-5f) + b[j];
  x[t] = v > 0.f ? v : 0.f;
}

// ---------------------------------------------------------------------------

extern "C" void kernel_launch(void* const* d_in, const int* in_sizes, int n_in,
                              void* d_out, int out_size, void* d_ws, size_t ws_size,
                              hipStream_t stream) {
  (void)in_sizes; (void)n_in; (void)out_size; (void)ws_size;
  // setup_inputs dict order: x, edge_index, edge_attr, batch, params(flattened, insertion order)
  const float* x      = (const float*)d_in[0];
  const int*   ei     = (const int*)  d_in[1];   // [2][E]: row0=src, row1=dst
  const float* ea     = (const float*)d_in[2];
  const int*   batch  = (const int*)  d_in[3];
  const float* node_w = (const float*)d_in[4];
  const float* node_b = (const float*)d_in[5];
  const float* edge_w = (const float*)d_in[6];
  const float* edge_b = (const float*)d_in[7];
  // convs[L]: edge_enc_w,b, pre_w,b, post_w,b, lin_w,b, bn_g, bn_b  (10 each)
  auto cp = [&](int L, int k) { return (const float*)d_in[8 + L * 10 + k]; };
  const float* mlp_w  = (const float*)d_in[38];
  const float* mlp_b  = (const float*)d_in[39];
  const float* mlp_g  = (const float*)d_in[40];
  const float* mlp_bb = (const float*)d_in[41];
  const float* o1_w = (const float*)d_in[42]; const float* o1_b  = (const float*)d_in[43];
  const float* o1_g = (const float*)d_in[44]; const float* o1_bb = (const float*)d_in[45];
  const float* o2_w = (const float*)d_in[46]; const float* o2_b  = (const float*)d_in[47];
  const float* o2_g = (const float*)d_in[48]; const float* o2_bb = (const float*)d_in[49];
  const float* o3_w = (const float*)d_in[50]; const float* o3_b  = (const float*)d_in[51];

  const int* srcI = ei;
  const int* dstI = ei + NE;

  // avg_log_deg from the compile-time DEG_HIST constant (host-side).
  const double hist[9] = {0, 1000, 5000, 10000, 15000, 10000, 5000, 2000, 500};
  double hs = 0, ht = 0;
  for (int i = 0; i < 9; ++i) { hs += log((double)i + 1.0) * hist[i]; ht += hist[i]; }
  float avgLog = (float)(hs / ht);

  // ---- workspace carve-up (≈200 MB) ----
  char* wsp = (char*)d_ws;
  auto alloc = [&](size_t bytes) { char* p = wsp; wsp += (bytes + 255) & ~(size_t)255; return p; };
  unsigned short* hbf   = (unsigned short*)alloc((size_t)NN * 64 * 2);
  unsigned short* ebf   = (unsigned short*)alloc((size_t)NE * 64 * 2);
  float* deg   = (float*)alloc((size_t)NN * 4);
  float* amp   = (float*)alloc((size_t)NN * 4);
  float* att   = (float*)alloc((size_t)NN * 4);
  float* sum   = (float*)alloc((size_t)NN * 64 * 4);
  float* ssq   = (float*)alloc((size_t)NN * 64 * 4);
  float* smn   = (float*)alloc((size_t)NN * 64 * 4);
  float* smx   = (float*)alloc((size_t)NN * 64 * 4);
  unsigned short* aggbf = (unsigned short*)alloc((size_t)NN * 256 * 2);
  float* outf  = (float*)alloc((size_t)NN * 64 * 4);
  float* Wef   = (float*)alloc(64 * 64 * 4);
  float* biasP = (float*)alloc(64 * 4);
  float* Pall  = (float*)alloc(832 * 64 * 4);
  float* b2    = (float*)alloc(64 * 4);
  unsigned short* packMsg = (unsigned short*)alloc(3 * 4096 * 2);   // [3][nt4][kt2][32][16]
  unsigned short* packPh  = (unsigned short*)alloc(4096 * 2);
  unsigned short* packP1  = (unsigned short*)alloc(16384 * 2);
  unsigned short* packP2  = (unsigned short*)alloc(16384 * 2);
  unsigned short* packP3  = (unsigned short*)alloc(16384 * 2);
  float* bnsum = (float*)alloc(64 * 4);
  float* bnsq  = (float*)alloc(64 * 4);
  float* pool  = (float*)alloc((size_t)NG * 64 * 4);
  float* tA    = (float*)alloc((size_t)NG * 64 * 4);
  float* tB    = (float*)alloc((size_t)NG * 64 * 4);
  float* hstat = (float*)alloc(128 * 4);

  const int T = 256;
  const int nf = NN * 64;                       // 3.2M
  const int nfB = (nf + T - 1) / T;

  // ---- encoders ----
  k_node_enc<<<(NN + T - 1) / T, T, 0, stream>>>(x, node_w, node_b, hbf);
  k_edge_enc<<<(NE + T - 1) / T, T, 0, stream>>>(ea, edge_w, edge_b, ebf);

  // ---- degree / amp / att ----
  (void)hipMemsetAsync(deg, 0, (size_t)NN * 4, stream);
  k_deg<<<(NE + T - 1) / T, T, 0, stream>>>(dstI, deg);
  k_ampatt<<<(NN + T - 1) / T, T, 0, stream>>>(deg, amp, att, avgLog);

  // ---- conv layers ----
  for (int L = 0; L < 3; ++L) {
    const float* encw = cp(L, 0); const float* encb = cp(L, 1);
    const float* prew = cp(L, 2); const float* preb = cp(L, 3);
    const float* postw = cp(L, 4); const float* postb = cp(L, 5);
    const float* linw = cp(L, 6); const float* linb = cp(L, 7);
    const float* bng = cp(L, 8); const float* bnb = cp(L, 9);

    k_compose_pre<<<16, T, 0, stream>>>(encw, encb, prew, preb, Wef, biasP);
    k_compose_post<<<(832 * 64 + T - 1) / T, T, 0, stream>>>(postw, postb, linw, linb, Pall, b2);

    k_pack_b<<<1, T, 0, stream>>>(prew, packMsg, 2);               // W1 = pre_w[0:64]
    k_pack_b<<<1, T, 0, stream>>>(prew + 64 * 64, packMsg + 4096, 2); // W2 = pre_w[64:128]
    k_pack_b<<<1, T, 0, stream>>>(Wef, packMsg + 8192, 2);         // We (composed)
    k_pack_b<<<1, T, 0, stream>>>(Pall, packPh, 2);                // P[0:64]
    k_pack_b<<<4, T, 0, stream>>>(Pall + 64 * 64, packP1, 8);      // P[64:320]
    k_pack_b<<<4, T, 0, stream>>>(Pall + 320 * 64, packP2, 8);     // P[320:576]
    k_pack_b<<<4, T, 0, stream>>>(Pall + 576 * 64, packP3, 8);     // P[576:832]

    (void)hipMemsetAsync(sum, 0, (size_t)nf * 4, stream);
    (void)hipMemsetAsync(ssq, 0, (size_t)nf * 4, stream);
    k_fill<<<nfB, T, 0, stream>>>(smn,  INFINITY, nf);
    k_fill<<<nfB, T, 0, stream>>>(smx, -INFINITY, nf);

    // 50000 waves x 16 edges, 8 waves/block
    k_msg<<<(NE / 16 + 7) / 8, 256, 0, stream>>>(hbf, ebf, srcI, dstI, packMsg, biasP,
                                                 sum, ssq, smn, smx);
    k_agg_fin<<<nfB, T, 0, stream>>>(sum, ssq, smn, smx, deg, aggbf);

    (void)hipMemsetAsync(bnsum, 0, 64 * 4, stream);
    (void)hipMemsetAsync(bnsq, 0, 64 * 4, stream);
    k_node_gemm<<<(NN / 16 + 7) / 8, 256, 0, stream>>>(hbf, aggbf, amp, att, packPh, packP1,
                                                       packP2, packP3, b2, outf, bnsum, bnsq);
    k_bn_apply<<<nfB, T, 0, stream>>>(outf, bnsum, bnsq, bng, bnb, hbf);
  }

  // ---- global add pool ----
  (void)hipMemsetAsync(pool, 0, (size_t)NG * 64 * 4, stream);
  k_pool<<<nfB, T, 0, stream>>>(hbf, batch, pool);

  // ---- head MLPs (tiny: plain VALU) ----
  const float* cur = pool;
  float* nxt = tA;
  for (int it = 0; it < 2; ++it) {
    (void)hipMemsetAsync(hstat, 0, 2 * 64 * 4, stream);
    k_head_gemm<<<(NG * 64 + T - 1) / T, T, 0, stream>>>(cur, mlp_w, mlp_b, nxt, 64, 64, hstat);
    k_head_bn<<<(NG * 64 + T - 1) / T, T, 0, stream>>>(nxt, mlp_g, mlp_bb, 64, hstat);
    cur = nxt; nxt = (nxt == tA) ? tB : tA;
  }
  (void)hipMemsetAsync(hstat, 0, 2 * 32 * 4, stream);
  k_head_gemm<<<(NG * 32 + T - 1) / T, T, 0, stream>>>(cur, o1_w, o1_b, nxt, 64, 32, hstat);
  k_head_bn<<<(NG * 32 + T - 1) / T, T, 0, stream>>>(nxt, o1_g, o1_bb, 32, hstat);
  cur = nxt; nxt = (nxt == tA) ? tB : tA;
  (void)hipMemsetAsync(hstat, 0, 2 * 16 * 4, stream);
  k_head_gemm<<<(NG * 16 + T - 1) / T, T, 0, stream>>>(cur, o2_w, o2_b, nxt, 32, 16, hstat);
  k_head_bn<<<(NG * 16 + T - 1) / T, T, 0, stream>>>(nxt, o2_g, o2_bb, 16, hstat);
  cur = nxt;
  k_head_gemm<<<(NG * 3 + T - 1) / T, T, 0, stream>>>(cur, o3_w, o3_b, (float*)d_out, 16, 3,
                                                      nullptr);
}
